// CrossAttention_21431886807769
// MI455X (gfx1250) — compile-verified, matrix operand order checked by reading, not run on hardware
//
#include <hip/hip_runtime.h>
#include <hip/hip_bf16.h>

typedef __bf16 bf16_t;
typedef unsigned int u32;
typedef __attribute__((ext_vector_type(16))) __bf16 v16bf;
typedef __attribute__((ext_vector_type(8)))  __bf16 v8bf;
typedef __attribute__((ext_vector_type(8)))  float  v8f;
typedef __attribute__((ext_vector_type(4)))  u32    u32x4;
typedef __attribute__((ext_vector_type(8)))  int    i32x8;
typedef __attribute__((ext_vector_type(4)))  int    i32x4;

constexpr int Dd  = 512;
constexpr int Bb  = 8;
constexpr int S1c = 2048;
constexpr int S2c = 2048;

// ln(10000)/256
#define ROPE_FREQ_K 0.035977892078032f
// 1/sqrt(512)
#define ATTN_SCALE 0.044194173824159216f

// LDS layout for attention kernel (bytes)
constexpr u32 SC_OFF  = 0;                        // f32 scores [16][2048] : 128K
constexpr u32 PB_OFF  = 16 * 2048 * 4;            // bf16 P     [16][2048] : 64K
constexpr u32 IST_OFF = PB_OFF + 16 * 2048 * 2;   // f32 1/sum  [16]
constexpr u32 KB_OFF  = IST_OFF + 256;            // K staging, 2 x 32K
constexpr u32 KB_SZ   = 32 * 512 * 2;             // one 32x512 bf16 tile
constexpr u32 ATTN_LDS = KB_OFF + 2 * KB_SZ;      // 262400 B < 320K

// LDS layout for projection kernels (bytes)
constexpr u32 PWL_OFF = 0;                        // W tile [64][512] bf16 : 64K
constexpr u32 PXL_OFF = 64 * 512 * 2;             // X tile [64][512] bf16 : 64K
constexpr u32 PROJ_LDS = PXL_OFF + 64 * 512 * 2;  // 128K

// ---------------------------------------------------------------------------
// Fragment loaders (CDNA5 16x16x32 bf16 WMMA layouts, wave32)
// A (16xK=32): lane L<16 -> row L, K {k..k+7, k+16..k+23}; lane L>=16 -> +8.
// B (K=32x16): same mapping on the transposed (n-major) operand.
// ---------------------------------------------------------------------------
__device__ __forceinline__ v16bf load_frag(const bf16_t* p0, int ld, int lane) {
  const int half = (lane & 16) >> 1;            // 0 or 8
  const bf16_t* p = p0 + (size_t)(lane & 15) * ld + half;
  union { v16bf v; struct { v8bf lo, hi; } s; } u;
  u.s.lo = *(const v8bf*)(p);
  u.s.hi = *(const v8bf*)(p + 16);
  return u.v;
}

__device__ __forceinline__ v8f wmma_bf16(v16bf a, v16bf b, v8f c) {
  return __builtin_amdgcn_wmma_f32_16x16x32_bf16(false, a, false, b,
                                                 (short)0, c, false, false);
}

// ---------------------------------------------------------------------------
// TDM: DMA `rows` x 512 bf16 row-major tile (row stride 512) global -> LDS.
// D# per CDNA5 ISA.
// ---------------------------------------------------------------------------
__device__ __forceinline__ void tdm_load_rows512(const bf16_t* g, u32 lds_off,
                                                 int rows) {
  const unsigned long long ga = (unsigned long long)(const void*)g;
  u32x4 g0;
  g0[0] = 1u;                                            // count=1 (valid)
  g0[1] = lds_off;                                       // lds_addr
  g0[2] = (u32)ga;                                       // global_addr[31:0]
  g0[3] = (u32)((ga >> 32) & 0x01ffffffULL) | 0x80000000u; // [56:32] | type=2
  i32x8 g1;
  g1[0] = (int)(1u << 16);          // workgroup_mask=0, data_size=1 (2 bytes)
  g1[1] = (int)(512u << 16);        // tensor_dim0 = 512 (lo16 -> word1 hi)
  g1[2] = (int)(8192u << 16);       // tensor_dim0 hi=0 | tensor_dim1 lo
  g1[3] = (int)(512u << 16);        // tensor_dim1 hi=0 | tile_dim0 = 512
  g1[4] = rows;                     // tile_dim1 = rows, tile_dim2 = 0
  g1[5] = 512;                      // tensor_dim0_stride = 512
  g1[6] = 0;
  g1[7] = 0;
  i32x4 z4 = {0, 0, 0, 0};
#if __clang_major__ >= 23
  i32x8 z8 = {0, 0, 0, 0, 0, 0, 0, 0};
  __builtin_amdgcn_tensor_load_to_lds(g0, g1, z4, z4, z8, 0);
#else
  __builtin_amdgcn_tensor_load_to_lds(g0, g1, z4, z4, 0);
#endif
}

// ---------------------------------------------------------------------------
// fp32 -> bf16 elementwise convert
// ---------------------------------------------------------------------------
__global__ void cvt_bf16_kernel(const float* __restrict__ x,
                                bf16_t* __restrict__ y, size_t n) {
  size_t i = (size_t)blockIdx.x * blockDim.x + threadIdx.x;
  size_t stride = (size_t)gridDim.x * blockDim.x;
  for (; i < n; i += stride) y[i] = (bf16_t)x[i];
}

// ---------------------------------------------------------------------------
// Q/K projection + RoPE.  Block = 4 waves, 64 rows; W tile (two 32-col
// ranges: [cb,cb+32) and [cb+256,cb+288)) and X tile staged in LDS via TDM.
// Wave w computes rows blockRow0 + w*16.
// ---------------------------------------------------------------------------
__global__ __launch_bounds__(128) void proj_rope_kernel(
    const bf16_t* __restrict__ X, const bf16_t* __restrict__ W,
    const float* __restrict__ bias, bf16_t* __restrict__ Out, int S) {
  extern __shared__ char psm[];
  bf16_t* Wl = (bf16_t*)(psm + PWL_OFF);   // rows 0..31: cols cb..cb+31
                                           // rows 32..63: cols cb+256..cb+287
  bf16_t* Xl = (bf16_t*)(psm + PXL_OFF);   // rows blockRow0..blockRow0+63

  const int lane = threadIdx.x & 31;
  const int wave = threadIdx.x >> 5;
  const int blockRow0 = blockIdx.x * 64;
  const int cb = blockIdx.y * 32;          // first-half column base

  // Each wave DMAs a 16-row quarter of W and of X into LDS.
  const int wsrc = (wave < 2) ? (cb + wave * 16) : (cb + 256 + (wave - 2) * 16);
  tdm_load_rows512(W + (size_t)wsrc * Dd, PWL_OFF + (u32)wave * 16 * 512 * 2, 16);
  tdm_load_rows512(X + (size_t)(blockRow0 + wave * 16) * Dd,
                   PXL_OFF + (u32)wave * 16 * 512 * 2, 16);
  __builtin_amdgcn_s_wait_tensorcnt(0);
  __syncthreads();

  const int row0 = blockRow0 + wave * 16;
  const bf16_t* Xw = Xl + (size_t)wave * 16 * 512;

  v8f acc[4] = {v8f{}, v8f{}, v8f{}, v8f{}};
#pragma unroll 4
  for (int kk = 0; kk < 16; ++kk) {
    const int k = kk * 32;
    v16bf a = load_frag(Xw + k, 512, lane);
#pragma unroll
    for (int t = 0; t < 4; ++t) {
      v16bf b = load_frag(Wl + (size_t)(t * 16) * 512 + k, 512, lane);
      acc[t] = wmma_bf16(a, b, acc[t]);
    }
  }

  const int s0 = row0 % S;                  // sequence position base
  const int mb = (lane & 16) ? 8 : 0;

#pragma unroll
  for (int pair = 0; pair < 2; ++pair) {
    const int c = cb + pair * 16 + (lane & 15);    // rope freq index 0..255
    const float invf = __expf(-(float)c * ROPE_FREQ_K);
    const float bv1 = bias[c];
    const float bv2 = bias[c + 256];
    v8f x1 = acc[pair];          // cols c       (Wl rows 0..31)
    v8f x2 = acc[pair + 2];      // cols c + 256 (Wl rows 32..63)
    bf16_t* o = Out + (size_t)row0 * Dd + c;
#pragma unroll
    for (int i = 0; i < 8; ++i) {
      const int m = mb + i;
      float sn, cs;
      __sincosf((float)(s0 + m) * invf, &sn, &cs);
      const float v1 = x1[i] + bv1;
      const float v2 = x2[i] + bv2;
      o[(size_t)m * Dd]       = (bf16_t)(v1 * cs - v2 * sn);
      o[(size_t)m * Dd + 256] = (bf16_t)(v1 * sn + v2 * cs);
    }
  }
}

// ---------------------------------------------------------------------------
// V projection, stored TRANSPOSED: Vt[b][d][s2].  W/X tiles TDM-staged.
// ---------------------------------------------------------------------------
__global__ __launch_bounds__(128) void proj_v_kernel(
    const bf16_t* __restrict__ X, const bf16_t* __restrict__ W,
    const float* __restrict__ bias, bf16_t* __restrict__ Vt, int S) {
  extern __shared__ char psm[];
  bf16_t* Wl = (bf16_t*)(psm + PWL_OFF);   // cols cb..cb+63
  bf16_t* Xl = (bf16_t*)(psm + PXL_OFF);

  const int lane = threadIdx.x & 31;
  const int wave = threadIdx.x >> 5;
  const int blockRow0 = blockIdx.x * 64;
  const int cb = blockIdx.y * 64;

  tdm_load_rows512(W + (size_t)(cb + wave * 16) * Dd,
                   PWL_OFF + (u32)wave * 16 * 512 * 2, 16);
  tdm_load_rows512(X + (size_t)(blockRow0 + wave * 16) * Dd,
                   PXL_OFF + (u32)wave * 16 * 512 * 2, 16);
  __builtin_amdgcn_s_wait_tensorcnt(0);
  __syncthreads();

  const int row0 = blockRow0 + wave * 16;
  const bf16_t* Xw = Xl + (size_t)wave * 16 * 512;

  v8f acc[4] = {v8f{}, v8f{}, v8f{}, v8f{}};
#pragma unroll 4
  for (int kk = 0; kk < 16; ++kk) {
    const int k = kk * 32;
    v16bf a = load_frag(Xw + k, 512, lane);
#pragma unroll
    for (int t = 0; t < 4; ++t) {
      v16bf b = load_frag(Wl + (size_t)(t * 16) * 512 + k, 512, lane);
      acc[t] = wmma_bf16(a, b, acc[t]);
    }
  }

  const int bidx = row0 / S;
  const int s0   = row0 % S;
  const int mb   = (lane & 16) ? 8 : 0;
#pragma unroll
  for (int t = 0; t < 4; ++t) {
    const int c = cb + t * 16 + (lane & 15);
    const float bv = bias[c];
    bf16_t* o = Vt + ((size_t)bidx * Dd + c) * S + s0 + mb;
#pragma unroll
    for (int i = 0; i < 8; ++i) o[i] = (bf16_t)(acc[t][i] + bv);
  }
}

// ---------------------------------------------------------------------------
// Output projection: fp32 result = attn(bf16) @ Wo^T + bo.  TDM-staged tiles.
// ---------------------------------------------------------------------------
__global__ __launch_bounds__(128) void proj_o_kernel(
    const bf16_t* __restrict__ X, const bf16_t* __restrict__ W,
    const float* __restrict__ bias, float* __restrict__ Out) {
  extern __shared__ char psm[];
  bf16_t* Wl = (bf16_t*)(psm + PWL_OFF);
  bf16_t* Xl = (bf16_t*)(psm + PXL_OFF);

  const int lane = threadIdx.x & 31;
  const int wave = threadIdx.x >> 5;
  const int blockRow0 = blockIdx.x * 64;
  const int cb = blockIdx.y * 64;

  tdm_load_rows512(W + (size_t)(cb + wave * 16) * Dd,
                   PWL_OFF + (u32)wave * 16 * 512 * 2, 16);
  tdm_load_rows512(X + (size_t)(blockRow0 + wave * 16) * Dd,
                   PXL_OFF + (u32)wave * 16 * 512 * 2, 16);
  __builtin_amdgcn_s_wait_tensorcnt(0);
  __syncthreads();

  const int row0 = blockRow0 + wave * 16;
  const bf16_t* Xw = Xl + (size_t)wave * 16 * 512;

  v8f acc[4] = {v8f{}, v8f{}, v8f{}, v8f{}};
#pragma unroll 4
  for (int kk = 0; kk < 16; ++kk) {
    const int k = kk * 32;
    v16bf a = load_frag(Xw + k, 512, lane);
#pragma unroll
    for (int t = 0; t < 4; ++t) {
      v16bf b = load_frag(Wl + (size_t)(t * 16) * 512 + k, 512, lane);
      acc[t] = wmma_bf16(a, b, acc[t]);
    }
  }

  const int mb = (lane & 16) ? 8 : 0;
#pragma unroll
  for (int t = 0; t < 4; ++t) {
    const int c = cb + t * 16 + (lane & 15);
    const float bv = bias[c];
#pragma unroll
    for (int i = 0; i < 8; ++i)
      Out[(size_t)(row0 + mb + i) * Dd + c] = acc[t][i] + bv;
  }
}

// ---------------------------------------------------------------------------
// Attention: one wave per 16 query rows of one batch.
//   phase 1: K tiles DMA'd to LDS by the TDM (double-buffered, TENSORcnt),
//            scores = Q K^T * scale -> f32 LDS [16][2048]
//   phase 2: row max / sum(exp); unnormalized exp -> bf16 LDS
//   phase 3: O = (P @ V) * (1/sum), software-pipelined V fragments
// ---------------------------------------------------------------------------
__global__ __launch_bounds__(32) void attn_kernel(
    const bf16_t* __restrict__ Q, const bf16_t* __restrict__ K,
    const bf16_t* __restrict__ Vt, bf16_t* __restrict__ Aout) {
  extern __shared__ char smem[];
  float*  sc  = (float*)(smem + SC_OFF);
  bf16_t* pb  = (bf16_t*)(smem + PB_OFF);
  float*  ist = (float*)(smem + IST_OFF);

  const int lane = threadIdx.x & 31;
  const int tile = blockIdx.x;            // B * (S1/16)
  const int b    = tile >> 7;
  const int q0   = (tile & 127) << 4;

  const bf16_t* Qb = Q  + ((size_t)b * S1c + q0) * Dd;
  const bf16_t* Kb = K  + (size_t)b * S2c * Dd;
  const bf16_t* Vb = Vt + (size_t)b * Dd * S2c;
  const int mb = (lane & 16) ? 8 : 0;

  // Preload all Q A-fragments for this 16-row tile (16 x v16bf).
  v16bf aq[16];
#pragma unroll
  for (int kk = 0; kk < 16; ++kk) aq[kk] = load_frag(Qb + kk * 32, Dd, lane);

  // ---- phase 1: scores, K staged through LDS by the TDM ----------------
  tdm_load_rows512(Kb, KB_OFF, 32);
  for (int t2 = 0; t2 < 64; ++t2) {
    const bf16_t* kt = (const bf16_t*)(smem + KB_OFF + (u32)(t2 & 1) * KB_SZ);
    if (t2 < 63) {
      // ensure prior reads of the other buffer retired before TDM overwrites
      asm volatile("s_wait_dscnt 0x0" ::: "memory");
      tdm_load_rows512(Kb + (size_t)(t2 + 1) * 32 * Dd,
                       KB_OFF + (u32)((t2 + 1) & 1) * KB_SZ, 32);
      __builtin_amdgcn_s_wait_tensorcnt(1);   // current buffer ready
    } else {
      __builtin_amdgcn_s_wait_tensorcnt(0);
    }
#pragma unroll
    for (int st = 0; st < 2; ++st) {
      v8f acc = {};
#pragma unroll
      for (int kk = 0; kk < 16; ++kk) {
        v16bf bk = load_frag(kt + (size_t)(st * 16) * Dd + kk * 32, Dd, lane);
        acc = wmma_bf16(aq[kk], bk, acc);
      }
      const int n = t2 * 32 + st * 16 + (lane & 15);
#pragma unroll
      for (int i = 0; i < 8; ++i)
        sc[(size_t)(mb + i) * 2048 + n] = acc[i] * ATTN_SCALE;
    }
  }
  __syncthreads();

  // ---- phase 2: softmax stats + unnormalized exp (bf16) ----------------
  for (int r = 0; r < 16; ++r) {
    float mx = -3.0e38f;
    for (int c = lane; c < 2048; c += 32) mx = fmaxf(mx, sc[(size_t)r * 2048 + c]);
#pragma unroll
    for (int o = 16; o >= 1; o >>= 1) mx = fmaxf(mx, __shfl_xor(mx, o, 32));
    float sm = 0.0f;
    for (int c = lane; c < 2048; c += 32) {
      float e = __expf(sc[(size_t)r * 2048 + c] - mx);
      pb[(size_t)r * 2048 + c] = (bf16_t)e;
      sm += e;
    }
#pragma unroll
    for (int o = 16; o >= 1; o >>= 1) sm += __shfl_xor(sm, o, 32);
    if (lane == 0) ist[r] = 1.0f / sm;
  }
  __syncthreads();

  // ---- phase 3: O = P @ V, 4 groups of 128 d-cols, pipelined -----------
  bf16_t* Ob = Aout + ((size_t)b * S1c + q0) * Dd;
  for (int dg = 0; dg < 4; ++dg) {
    v8f acc[8] = {v8f{}, v8f{}, v8f{}, v8f{}, v8f{}, v8f{}, v8f{}, v8f{}};
    const bf16_t* Vg = Vb + (size_t)(dg * 128) * S2c;

    v16bf ap = load_frag(pb, 2048, lane);
    v16bf bv[8];
#pragma unroll
    for (int j = 0; j < 8; ++j)
      bv[j] = load_frag(Vg + (size_t)(j * 16) * S2c, S2c, lane);

    for (int s2k = 0; s2k < 63; ++s2k) {
      const int kn = (s2k + 1) * 32;
      v16bf apn = load_frag(pb + kn, 2048, lane);
      v16bf bvn[8];
#pragma unroll
      for (int j = 0; j < 8; ++j)
        bvn[j] = load_frag(Vg + (size_t)(j * 16) * S2c + kn, S2c, lane);
#pragma unroll
      for (int j = 0; j < 8; ++j) acc[j] = wmma_bf16(ap, bv[j], acc[j]);
      ap = apn;
#pragma unroll
      for (int j = 0; j < 8; ++j) bv[j] = bvn[j];
    }
#pragma unroll
    for (int j = 0; j < 8; ++j) acc[j] = wmma_bf16(ap, bv[j], acc[j]);

#pragma unroll
    for (int j = 0; j < 8; ++j) {
      const int dcol = dg * 128 + j * 16 + (lane & 15);
#pragma unroll
      for (int i = 0; i < 8; ++i) {
        const float w = ist[mb + i];
        Ob[(size_t)(mb + i) * Dd + dcol] = (bf16_t)(acc[j][i] * w);
      }
    }
  }
}

// ---------------------------------------------------------------------------
// Host launcher
// ---------------------------------------------------------------------------
extern "C" void kernel_launch(void* const* d_in, const int* in_sizes, int n_in,
                              void* d_out, int out_size, void* d_ws, size_t ws_size,
                              hipStream_t stream) {
  (void)in_sizes; (void)n_in; (void)out_size; (void)ws_size;

  const float* h1 = (const float*)d_in[0];
  const float* h2 = (const float*)d_in[1];
  const float* Wq = (const float*)d_in[2];
  const float* bq = (const float*)d_in[3];
  const float* Wk = (const float*)d_in[4];
  const float* bk = (const float*)d_in[5];
  const float* Wv = (const float*)d_in[6];
  const float* bv = (const float*)d_in[7];
  const float* Wo = (const float*)d_in[8];
  const float* bo = (const float*)d_in[9];
  float* out = (float*)d_out;

  char* w = (char*)d_ws;
  size_t off = 0;
  auto alloc = [&](size_t bytes) -> char* {
    char* p = w + off;
    off += (bytes + 255) & ~(size_t)255;
    return p;
  };
  const size_t nh1 = (size_t)Bb * S1c * Dd;
  const size_t nh2 = (size_t)Bb * S2c * Dd;
  const size_t nw  = (size_t)Dd * Dd;

  bf16_t* h1b = (bf16_t*)alloc(nh1 * 2);
  bf16_t* h2b = (bf16_t*)alloc(nh2 * 2);
  bf16_t* Wqb = (bf16_t*)alloc(nw * 2);
  bf16_t* Wkb = (bf16_t*)alloc(nw * 2);
  bf16_t* Wvb = (bf16_t*)alloc(nw * 2);
  bf16_t* Wob = (bf16_t*)alloc(nw * 2);
  bf16_t* Qb  = (bf16_t*)alloc(nh1 * 2);
  bf16_t* Kb  = (bf16_t*)alloc(nh2 * 2);
  bf16_t* Vtb = (bf16_t*)alloc(nh2 * 2);
  bf16_t* Ab  = (bf16_t*)alloc(nh1 * 2);

  // bf16 conversions
  cvt_bf16_kernel<<<2048, 256, 0, stream>>>(h1, h1b, nh1);
  cvt_bf16_kernel<<<2048, 256, 0, stream>>>(h2, h2b, nh2);
  cvt_bf16_kernel<<<256, 256, 0, stream>>>(Wq, Wqb, nw);
  cvt_bf16_kernel<<<256, 256, 0, stream>>>(Wk, Wkb, nw);
  cvt_bf16_kernel<<<256, 256, 0, stream>>>(Wv, Wvb, nw);
  cvt_bf16_kernel<<<256, 256, 0, stream>>>(Wo, Wob, nw);

  // Q/K projections with fused RoPE (W/X tiles TDM-staged in LDS)
  dim3 gQK((unsigned)(Bb * S1c / 64), 8);
  proj_rope_kernel<<<gQK, 128, PROJ_LDS, stream>>>(h1b, Wqb, bq, Qb, S1c);
  dim3 gK((unsigned)(Bb * S2c / 64), 8);
  proj_rope_kernel<<<gK, 128, PROJ_LDS, stream>>>(h2b, Wkb, bk, Kb, S2c);

  // V projection (transposed output)
  dim3 gV((unsigned)(Bb * S2c / 64), 8);
  proj_v_kernel<<<gV, 128, PROJ_LDS, stream>>>(h2b, Wvb, bv, Vtb, S2c);

  // attention (TDM-staged K, double-buffered)
  attn_kernel<<<(unsigned)(Bb * S1c / 16), 32, ATTN_LDS, stream>>>(Qb, Kb, Vtb, Ab);

  // output projection
  dim3 gO((unsigned)(Bb * S1c / 64), 8);
  proj_o_kernel<<<gO, 128, PROJ_LDS, stream>>>(Ab, Wob, bo, out);
}
